// GAT_69793218560235
// MI455X (gfx1250) — matrix operand order repaired and checked by reading, not compiled, not run on hardware
//
#include <hip/hip_runtime.h>
#include <hip/hip_bf16.h>
#include <cstdint>

// ---------------------------------------------------------------------------
// Types for gfx1250 WMMA
// ---------------------------------------------------------------------------
typedef __bf16 bf16_t;
typedef __attribute__((ext_vector_type(16))) __bf16 v16bf;
typedef __attribute__((ext_vector_type(8)))  __bf16 v8bf;
typedef __attribute__((ext_vector_type(8)))  float  v8f;

// ---------------------------------------------------------------------------
// Utility kernels
// ---------------------------------------------------------------------------
__global__ void gat_fill_f32(float* __restrict__ p, float v, long long n) {
    long long i = (long long)blockIdx.x * blockDim.x + threadIdx.x;
    long long step = (long long)gridDim.x * blockDim.x;
    for (; i < n; i += step) p[i] = v;
}

__global__ void gat_f32_to_bf16(const float* __restrict__ in, bf16_t* __restrict__ out,
                                long long n) {
    long long i = (long long)blockIdx.x * blockDim.x + threadIdx.x;
    long long step = (long long)gridDim.x * blockDim.x;
    for (; i < n; i += step) out[i] = (bf16_t)in[i];
}

// W [K, Ncol] row-major (f32)  ->  Wt [Ncol, K] row-major (bf16)
__global__ void gat_transpose_bf16(const float* __restrict__ W, bf16_t* __restrict__ Wt,
                                   int K, int Ncol) {
    long long n = (long long)K * Ncol;
    long long i = (long long)blockIdx.x * blockDim.x + threadIdx.x;
    long long step = (long long)gridDim.x * blockDim.x;
    for (; i < n; i += step) {
        int k = (int)(i / Ncol);
        int c = (int)(i % Ncol);
        Wt[(long long)c * K + k] = (bf16_t)W[i];
    }
}

// ---------------------------------------------------------------------------
// WMMA GEMM: C[M,Ncol] = A[M,K] (bf16 row-major) x Bt[Ncol,K]^T (bf16)
// One wave per 16x64 output strip (4 accumulators): the A fragment is loaded
// once per K-step and reused by 4 WMMAs -> ~2x arithmetic intensity vs 16x16.
// M multiple of 16, Ncol multiple of 64, K multiple of 32.
// ---------------------------------------------------------------------------
__global__ void gat_gemm_bf16_wmma(const bf16_t* __restrict__ A,
                                   const bf16_t* __restrict__ Bt,
                                   float* __restrict__ C,
                                   int M, int Ncol, int K) {
    const int lane = threadIdx.x & 31;
    const int wave = threadIdx.x >> 5;
    const int m0 = (blockIdx.x * 8 + wave) * 16;
    const int n0 = blockIdx.y * 64;
    if (m0 >= M) return;                       // wave-uniform: EXEC stays all-ones

    const int rowA = m0 + (lane & 15);
    const int kSubA = (lane >> 4) * 8;         // A: lanes 16-31 start at K+8
    const int kSubB = (lane >> 4) * 8;         // B: lanes 16-31 start at K+8

    const bf16_t* __restrict__ aRow = A + (size_t)rowA * K;
    const bf16_t* __restrict__ bRow0 = Bt + (size_t)(n0 +  0 + (lane & 15)) * K;
    const bf16_t* __restrict__ bRow1 = Bt + (size_t)(n0 + 16 + (lane & 15)) * K;
    const bf16_t* __restrict__ bRow2 = Bt + (size_t)(n0 + 32 + (lane & 15)) * K;
    const bf16_t* __restrict__ bRow3 = Bt + (size_t)(n0 + 48 + (lane & 15)) * K;

    v8f acc0 = {}, acc1 = {}, acc2 = {}, acc3 = {};
    for (int k0 = 0; k0 < K; k0 += 32) {
        // A fragment: elems 0..7 -> K = k0+kSubA+[0..7], elems 8..15 -> +16
        v8bf a0 = *(const v8bf*)(aRow + k0 + kSubA);
        v8bf a1 = *(const v8bf*)(aRow + k0 + kSubA + 16);
        v16bf a;
#pragma unroll
        for (int i = 0; i < 8; ++i) { a[i] = a0[i]; a[i + 8] = a1[i]; }

        // B fragments: elems 0..7 -> K = k0+kSubB+[0..7], elems 8..15 -> +16
        const bf16_t* brs[4] = { bRow0, bRow1, bRow2, bRow3 };
        v16bf b[4];
#pragma unroll
        for (int t = 0; t < 4; ++t) {
            v8bf blo = *(const v8bf*)(brs[t] + k0 + kSubB);
            v8bf bhi = *(const v8bf*)(brs[t] + k0 + kSubB + 16);
#pragma unroll
            for (int i = 0; i < 8; ++i) { b[t][i] = blo[i]; b[t][i + 8] = bhi[i]; }
        }

        acc0 = __builtin_amdgcn_wmma_f32_16x16x32_bf16(false, a, false, b[0], (short)0, acc0, false, false);
        acc1 = __builtin_amdgcn_wmma_f32_16x16x32_bf16(false, a, false, b[1], (short)0, acc1, false, false);
        acc2 = __builtin_amdgcn_wmma_f32_16x16x32_bf16(false, a, false, b[2], (short)0, acc2, false, false);
        acc3 = __builtin_amdgcn_wmma_f32_16x16x32_bf16(false, a, false, b[3], (short)0, acc3, false, false);
    }

    // C/D layout: lane L -> N = L%16 ; VGPR r -> M = r + (L/16)*8
    const int nLane = lane & 15;
    const int mBase = m0 + (lane >> 4) * 8;
#pragma unroll
    for (int r = 0; r < 8; ++r) {
        float* crow = C + (size_t)(mBase + r) * Ncol + n0 + nLane;
        crow[0]  = acc0[r];
        crow[16] = acc1[r];
        crow[32] = acc2[r];
        crow[48] = acc3[r];
    }
}

// ---------------------------------------------------------------------------
// Attention scores: a_src[n,h] = sum_c h[n,h,c]*att_src[h,c] (same for dst)
// ---------------------------------------------------------------------------
__global__ void gat_attn_scores(const float* __restrict__ h,
                                const float* __restrict__ att_s,
                                const float* __restrict__ att_d,
                                float* __restrict__ a_src,
                                float* __restrict__ a_dst,
                                int Nn, int heads, int ch) {
    long long tot = (long long)Nn * heads;
    long long i = (long long)blockIdx.x * blockDim.x + threadIdx.x;
    long long step = (long long)gridDim.x * blockDim.x;
    for (; i < tot; i += step) {
        int n  = (int)(i / heads);
        int hd = (int)(i % heads);
        const float* hp = h + ((size_t)n * heads + hd) * ch;
        const float* sp = att_s + (size_t)hd * ch;
        const float* dp = att_d + (size_t)hd * ch;
        float s1 = 0.f, s2 = 0.f;
        for (int c = 0; c < ch; ++c) { s1 += hp[c] * sp[c]; s2 += hp[c] * dp[c]; }
        a_src[i] = s1;
        a_dst[i] = s2;
    }
}

// ---------------------------------------------------------------------------
// Edge helpers (edges 0..E-1 from edge_index, E..E+N-1 are self-loops)
// ---------------------------------------------------------------------------
__device__ __forceinline__ void gat_edge_ep(long long e, const long long* __restrict__ ei,
                                            long long E, int& s, int& d) {
    if (e < E) { s = (int)ei[e]; d = (int)ei[E + e]; }
    else       { s = d = (int)(e - E); }
}

__device__ __forceinline__ void gat_atomic_max_f32(float* addr, float val) {
    unsigned int* ua = (unsigned int*)addr;
    unsigned int cur = *ua;
    while (__uint_as_float(cur) < val) {
        unsigned int prev = atomicCAS(ua, cur, __float_as_uint(val));
        if (prev == cur) break;
        cur = prev;
    }
}

__global__ void gat_edge_max(const long long* __restrict__ ei, long long E, long long Etot,
                             const float* __restrict__ a_src, const float* __restrict__ a_dst,
                             float* __restrict__ mmax, int heads) {
    long long tot = Etot * heads;
    long long i = (long long)blockIdx.x * blockDim.x + threadIdx.x;
    long long step = (long long)gridDim.x * blockDim.x;
    for (; i < tot; i += step) {
        long long e = i / heads;
        int hd = (int)(i % heads);
        int s, d;
        gat_edge_ep(e, ei, E, s, d);
        float v = a_src[(size_t)s * heads + hd] + a_dst[(size_t)d * heads + hd];
        v = v > 0.f ? v : 0.2f * v;            // leaky_relu(0.2)
        gat_atomic_max_f32(&mmax[(size_t)d * heads + hd], v);
    }
}

__global__ void gat_edge_exp(const long long* __restrict__ ei, long long E, long long Etot,
                             const float* __restrict__ a_src, const float* __restrict__ a_dst,
                             const float* __restrict__ mmax,
                             float* __restrict__ alpha, float* __restrict__ denom, int heads) {
    long long tot = Etot * heads;
    long long i = (long long)blockIdx.x * blockDim.x + threadIdx.x;
    long long step = (long long)gridDim.x * blockDim.x;
    for (; i < tot; i += step) {
        long long e = i / heads;
        int hd = (int)(i % heads);
        int s, d;
        gat_edge_ep(e, ei, E, s, d);
        float v = a_src[(size_t)s * heads + hd] + a_dst[(size_t)d * heads + hd];
        v = v > 0.f ? v : 0.2f * v;
        float ex = __expf(v - mmax[(size_t)d * heads + hd]);
        alpha[i] = ex;
        atomicAdd(&denom[(size_t)d * heads + hd], ex);
    }
}

__global__ void gat_edge_norm(const long long* __restrict__ ei, long long E, long long Etot,
                              const float* __restrict__ denom,
                              float* __restrict__ alpha, int heads) {
    long long tot = Etot * heads;
    long long i = (long long)blockIdx.x * blockDim.x + threadIdx.x;
    long long step = (long long)gridDim.x * blockDim.x;
    for (; i < tot; i += step) {
        long long e = i / heads;
        int hd = (int)(i % heads);
        int s, d;
        gat_edge_ep(e, ei, E, s, d);
        alpha[i] /= (denom[(size_t)d * heads + hd] + 1e-16f);
    }
}

// out[dst,h,c] += h[src,h,c] * alpha[e,h]   (4 channels per thread)
__global__ void gat_aggregate(const long long* __restrict__ ei, long long E, long long Etot,
                              const float* __restrict__ h, const float* __restrict__ alpha,
                              float* __restrict__ out, int heads, int ch) {
    int chunks = ch >> 2;
    long long tot = Etot * heads * chunks;
    long long i = (long long)blockIdx.x * blockDim.x + threadIdx.x;
    long long step = (long long)gridDim.x * blockDim.x;
    for (; i < tot; i += step) {
        long long e = i / ((long long)heads * chunks);
        int rem = (int)(i % ((long long)heads * chunks));
        int hd = rem / chunks;
        int c4 = (rem % chunks) * 4;
        int s, d;
        gat_edge_ep(e, ei, E, s, d);
        float al = alpha[e * heads + hd];
        const float4 hv = *(const float4*)(h + ((size_t)s * heads + hd) * ch + c4);
        float* op = out + ((size_t)d * heads + hd) * ch + c4;
        atomicAdd(op + 0, hv.x * al);
        atomicAdd(op + 1, hv.y * al);
        atomicAdd(op + 2, hv.z * al);
        atomicAdd(op + 3, hv.w * al);
    }
}

// out += bias ; optional ELU ; optional bf16 copy for next layer's GEMM input
__global__ void gat_bias_act(float* __restrict__ out, const float* __restrict__ bias,
                             bf16_t* __restrict__ nxt, int HC, long long total, int do_elu) {
    long long i = (long long)blockIdx.x * blockDim.x + threadIdx.x;
    long long step = (long long)gridDim.x * blockDim.x;
    for (; i < total; i += step) {
        int j = (int)(i % HC);
        float v = out[i] + bias[j];
        if (do_elu) v = v > 0.f ? v : (__expf(v) - 1.f);
        out[i] = v;
        if (nxt) nxt[i] = (bf16_t)v;
    }
}

// ---------------------------------------------------------------------------
// Global mean pool + final linear + double-sigmoid + BCE loss
// ---------------------------------------------------------------------------
__global__ void gat_pool_accum(const float* __restrict__ out3, const long long* __restrict__ batch,
                               float* __restrict__ sums, float* __restrict__ cnt,
                               int Nn, int HC) {
    long long tot = (long long)Nn * HC;
    long long i = (long long)blockIdx.x * blockDim.x + threadIdx.x;
    long long step = (long long)gridDim.x * blockDim.x;
    for (; i < tot; i += step) {
        int n = (int)(i / HC);
        int j = (int)(i % HC);
        int g = (int)batch[n];
        atomicAdd(&sums[(size_t)g * HC + j], out3[i]);
        if (j == 0) atomicAdd(&cnt[g], 1.0f);
    }
}

__global__ void gat_final(const float* __restrict__ sums, const float* __restrict__ cnt,
                          const float* __restrict__ Wl, const float* __restrict__ bl,
                          const int* __restrict__ y, float* __restrict__ d_out,
                          int G, int HC) {
    __shared__ float red[256];
    int g = threadIdx.x;
    float term = 0.f;
    if (g < G) {
        float c = fmaxf(cnt[g], 1.0f);
        float logit = 0.f;
        const float* sp = sums + (size_t)g * HC;
        for (int j = 0; j < HC; ++j) {
            float p = fmaxf(sp[j] / c, 0.f);   // relu(mean)
            logit += p * Wl[j];
        }
        logit += bl[0];
        float s1 = 1.f / (1.f + __expf(-logit));
        d_out[g] = 1.f / (1.f + __expf(-s1));  // sigmoid(sigmoid(logit))
        float yf = (float)y[g];
        term = fmaxf(logit, 0.f) - logit * yf + log1pf(__expf(-fabsf(logit)));
    }
    red[threadIdx.x] = term;
    __syncthreads();
    for (int s = 128; s > 0; s >>= 1) {
        if (threadIdx.x < s) red[threadIdx.x] += red[threadIdx.x + s];
        __syncthreads();
    }
    if (threadIdx.x == 0) d_out[G] = red[0] / (float)G;
}

// ---------------------------------------------------------------------------
// Host launcher
// ---------------------------------------------------------------------------
static inline int gat_blocks(long long total, int tpb = 256) {
    long long b = (total + tpb - 1) / tpb;
    if (b > (1ll << 20)) b = 1ll << 20;
    if (b < 1) b = 1;
    return (int)b;
}

extern "C" void kernel_launch(void* const* d_in, const int* in_sizes, int n_in,
                              void* d_out, int out_size, void* d_ws, size_t ws_size,
                              hipStream_t stream) {
    // Inputs (setup_inputs order)
    const float*     x    = (const float*)d_in[0];
    const int*       y    = (const int*)d_in[1];
    const long long* ei   = (const long long*)d_in[2];
    const long long* batch= (const long long*)d_in[3];
    const float* W1  = (const float*)d_in[4];
    const float* as1 = (const float*)d_in[5];
    const float* ad1 = (const float*)d_in[6];
    const float* b1  = (const float*)d_in[7];
    const float* W2  = (const float*)d_in[8];
    const float* as2 = (const float*)d_in[9];
    const float* ad2 = (const float*)d_in[10];
    const float* b2  = (const float*)d_in[11];
    const float* W3  = (const float*)d_in[12];
    const float* as3 = (const float*)d_in[13];
    const float* ad3 = (const float*)d_in[14];
    const float* b3  = (const float*)d_in[15];
    const float* Wl  = (const float*)d_in[16];
    const float* bl  = (const float*)d_in[17];

    const int       N    = in_sizes[3];          // 30000
    const long long E    = in_sizes[2] / 2;      // 480000
    const int       G    = in_sizes[1];          // 256
    const long long Etot = E + N;
    const int HDS = 4;
    const int K1 = 128, HC1 = 512, C1 = 128;
    const int K2 = 512, HC2 = 256, C2 = 64;
    const int K3 = 256, HC3 = 128, C3 = 32;

    // ---- workspace layout (256B aligned) ----
    char* base = (char*)d_ws;
    size_t off = 0;
    auto arena = [&](size_t bytes) -> char* {
        char* p = base + off;
        off += (bytes + 255) & ~(size_t)255;
        return p;
    };
    bf16_t* bufA  = (bf16_t*)arena((size_t)N * 512 * sizeof(bf16_t)); // gemm input (bf16)
    float*  bufH  = (float*) arena((size_t)N * 512 * sizeof(float));  // gemm output h
    float*  bufO  = (float*) arena((size_t)N * 512 * sizeof(float));  // aggregated out
    bf16_t* Wt    = (bf16_t*)arena((size_t)512 * 512 * sizeof(bf16_t));
    float*  a_src = (float*) arena((size_t)N * HDS * sizeof(float));
    float*  a_dst = (float*) arena((size_t)N * HDS * sizeof(float));
    float*  mmax  = (float*) arena((size_t)N * HDS * sizeof(float));
    float*  denom = (float*) arena((size_t)N * HDS * sizeof(float));
    float*  alpha = (float*) arena((size_t)Etot * HDS * sizeof(float));
    float*  sums  = (float*) arena((size_t)G * HC3 * sizeof(float));
    float*  cnt   = (float*) arena((size_t)G * sizeof(float));

    const int Mtiles = N / 16;                   // 1875 (N multiple of 16)

    auto run_layer = [&](const bf16_t* inb, int K, const float* W,
                         const float* as_, const float* ad_, const float* bias,
                         int HC, int ch, bf16_t* nxt, int do_elu) {
        // weight transpose + bf16
        gat_transpose_bf16<<<gat_blocks((long long)K * HC), 256, 0, stream>>>(W, Wt, K, HC);
        // GEMM h = inb @ W  via WMMA bf16 (16x64 strip per wave)
        dim3 grid((Mtiles + 7) / 8, HC / 64);
        gat_gemm_bf16_wmma<<<grid, 256, 0, stream>>>(inb, Wt, bufH, N, HC, K);
        // attention scores
        gat_attn_scores<<<gat_blocks((long long)N * HDS), 256, 0, stream>>>(
            bufH, as_, ad_, a_src, a_dst, N, HDS, ch);
        // init
        gat_fill_f32<<<gat_blocks((long long)N * HDS), 256, 0, stream>>>(mmax, -1e30f, (long long)N * HDS);
        gat_fill_f32<<<gat_blocks((long long)N * HDS), 256, 0, stream>>>(denom, 0.f, (long long)N * HDS);
        gat_fill_f32<<<gat_blocks((long long)N * HC), 256, 0, stream>>>(bufO, 0.f, (long long)N * HC);
        // segment softmax
        gat_edge_max<<<gat_blocks(Etot * HDS), 256, 0, stream>>>(ei, E, Etot, a_src, a_dst, mmax, HDS);
        gat_edge_exp<<<gat_blocks(Etot * HDS), 256, 0, stream>>>(ei, E, Etot, a_src, a_dst, mmax,
                                                                 alpha, denom, HDS);
        gat_edge_norm<<<gat_blocks(Etot * HDS), 256, 0, stream>>>(ei, E, Etot, denom, alpha, HDS);
        // aggregation
        gat_aggregate<<<gat_blocks(Etot * HDS * (ch / 4)), 256, 0, stream>>>(
            ei, E, Etot, bufH, alpha, bufO, HDS, ch);
        // bias (+elu) (+bf16 for next layer)
        gat_bias_act<<<gat_blocks((long long)N * HC), 256, 0, stream>>>(
            bufO, bias, nxt, HC, (long long)N * HC, do_elu);
    };

    // layer 1: x -> bf16, then GAT
    gat_f32_to_bf16<<<gat_blocks((long long)N * K1), 256, 0, stream>>>(x, bufA, (long long)N * K1);
    run_layer(bufA, K1, W1, as1, ad1, b1, HC1, C1, bufA, 1);   // writes next input into bufA
    run_layer(bufA, K2, W2, as2, ad2, b2, HC2, C2, bufA, 1);
    run_layer(bufA, K3, W3, as3, ad3, b3, HC3, C3, nullptr, 0); // out3 left in bufO (f32)

    // global mean pool + head
    gat_fill_f32<<<gat_blocks((long long)G * HC3), 256, 0, stream>>>(sums, 0.f, (long long)G * HC3);
    gat_fill_f32<<<gat_blocks((long long)G), 256, 0, stream>>>(cnt, 0.f, (long long)G);
    gat_pool_accum<<<gat_blocks((long long)N * HC3), 256, 0, stream>>>(bufO, batch, sums, cnt, N, HC3);
    gat_final<<<1, 256, 0, stream>>>(sums, cnt, Wl, bl, y, (float*)d_out, G, HC3);
}
